// FPQuantizedLinear_73684458930278
// MI455X (gfx1250) — 8-bit matrix operand order repaired and checked by reading, compiled, not run on hardware
//
#include <hip/hip_runtime.h>


typedef int v8i __attribute__((ext_vector_type(8)));
typedef int v4i __attribute__((ext_vector_type(4)));
typedef unsigned long long u64;

#define M_DIM 16384
#define N_DIM 4096
#define K_DIM 4096

#if __has_builtin(__builtin_amdgcn_global_load_async_to_lds_b128)
#define HAVE_ASYNC 1
#else
#define HAVE_ASYNC 0
#endif

__device__ __forceinline__ void wait_async_le6() {
#if HAVE_ASYNC
#if __has_builtin(__builtin_amdgcn_s_wait_asynccnt)
    __builtin_amdgcn_s_wait_asynccnt(6);
#else
    asm volatile("s_wait_asynccnt 0x6" ::: "memory");
#endif
#endif
}
__device__ __forceinline__ void wait_async_le0() {
#if HAVE_ASYNC
#if __has_builtin(__builtin_amdgcn_s_wait_asynccnt)
    __builtin_amdgcn_s_wait_asynccnt(0);
#else
    asm volatile("s_wait_asynccnt 0x0" ::: "memory");
#endif
#endif
}

__device__ __forceinline__ int quant_code(float v) {
    float q = rintf(v * 256.0f);
    q = fminf(fmaxf(q, -32768.0f), 32767.0f);
    return (int)q;
}

__global__ void quant_split_kernel(const float* __restrict__ in,
                                   unsigned char* __restrict__ hi,
                                   unsigned char* __restrict__ lo,
                                   long long n4) {
    long long i = (long long)blockIdx.x * blockDim.x + threadIdx.x;
    if (i >= n4) return;
    const float4 v = reinterpret_cast<const float4*>(in)[i];
    int c0 = quant_code(v.x), c1 = quant_code(v.y);
    int c2 = quant_code(v.z), c3 = quant_code(v.w);
    unsigned hp = ((unsigned)((c0 >> 8) & 0xff))       |
                  ((unsigned)((c1 >> 8) & 0xff) << 8)  |
                  ((unsigned)((c2 >> 8) & 0xff) << 16) |
                  ((unsigned)((c3 >> 8) & 0xff) << 24);
    unsigned lp = ((unsigned)(c0 & 0xff))       |
                  ((unsigned)(c1 & 0xff) << 8)  |
                  ((unsigned)(c2 & 0xff) << 16) |
                  ((unsigned)(c3 & 0xff) << 24);
    reinterpret_cast<unsigned*>(hi)[i] = hp;
    reinterpret_cast<unsigned*>(lo)[i] = lp;
}

__device__ __forceinline__ v8i make_frag(u64 q0, u64 q1, u64 q2, u64 q3) {
    v8i f;
    f[0] = (int)q0; f[1] = (int)(q0 >> 32);
    f[2] = (int)q1; f[3] = (int)(q1 >> 32);
    f[4] = (int)q2; f[5] = (int)(q2 >> 32);
    f[6] = (int)q3; f[7] = (int)(q3 >> 32);
    return f;
}

__global__ __launch_bounds__(256, 1)
void fpq_gemm_kernel(const unsigned char* __restrict__ xh,
                     const unsigned char* __restrict__ xl,
                     const unsigned char* __restrict__ wh,
                     const unsigned char* __restrict__ wl,
                     const float* __restrict__ bias,
                     float* __restrict__ out) {
    __shared__ u64 sAh[3][128 * 8];
    __shared__ u64 sAl[3][128 * 8];
    __shared__ u64 sBh[3][64 * 8];
    __shared__ u64 sBl[3][64 * 8];

    const int t    = threadIdx.x;
    const int m0   = blockIdx.x * 128;
    const int n0   = blockIdx.y * 64;
    const int lane = t & 31;
    const int wid  = t >> 5;
    const int wm   = (wid >> 1) * 32;
    const int wn   = (wid & 1) * 32;
    const int lr   = lane & 15;
    const int lh   = lane >> 4;

#if HAVE_ASYNC
    auto async_copy = [&](int buf, int kk) {
#pragma unroll
        for (int i = 0; i < 2; ++i) {
            int lin = t + 256 * i;
            int row = lin >> 2, c = lin & 3;
            size_t goff = (size_t)(m0 + row) * K_DIM + (size_t)kk + (size_t)c * 16;
            unsigned char* la = (unsigned char*)&sAh[buf][0] + lin * 16;
            unsigned char* lb = (unsigned char*)&sAl[buf][0] + lin * 16;
            __builtin_amdgcn_global_load_async_to_lds_b128(
                (v4i*)(xh + goff), (v4i*)la, 0, 0);
            __builtin_amdgcn_global_load_async_to_lds_b128(
                (v4i*)(xl + goff), (v4i*)lb, 0, 0);
        }
        {
            int row = t >> 2, c = t & 3;
            size_t goff = (size_t)(n0 + row) * K_DIM + (size_t)kk + (size_t)c * 16;
            unsigned char* la = (unsigned char*)&sBh[buf][0] + t * 16;
            unsigned char* lb = (unsigned char*)&sBl[buf][0] + t * 16;
            __builtin_amdgcn_global_load_async_to_lds_b128(
                (v4i*)(wh + goff), (v4i*)la, 0, 0);
            __builtin_amdgcn_global_load_async_to_lds_b128(
                (v4i*)(wl + goff), (v4i*)lb, 0, 0);
        }
    };
#endif

    const v8i vzero = {0, 0, 0, 0, 0, 0, 0, 0};
    v8i acc_hh[2][2], acc_mm[2][2], acc_ll[2][2];
#pragma unroll
    for (int mi = 0; mi < 2; ++mi)
#pragma unroll
        for (int ni = 0; ni < 2; ++ni) {
            acc_hh[mi][ni] = vzero;
            acc_mm[mi][ni] = vzero;
            acc_ll[mi][ni] = vzero;
        }

    auto compute = [&](int cur) {
        v8i ah[2], al[2], bh[2], bl[2];
#pragma unroll
        for (int mi = 0; mi < 2; ++mi) {
            const int row = wm + 16 * mi + lr;
            const u64* ph = &sAh[cur][row * 8];
            const u64* pl = &sAl[cur][row * 8];
            ah[mi] = make_frag(ph[lh], ph[2 + lh], ph[4 + lh], ph[6 + lh]);
            al[mi] = make_frag(pl[lh], pl[2 + lh], pl[4 + lh], pl[6 + lh]);
        }
#pragma unroll
        for (int ni = 0; ni < 2; ++ni) {
            const int nl = wn + 16 * ni + lr;
            const u64* ph = &sBh[cur][nl * 8];
            const u64* pl = &sBl[cur][nl * 8];
            bh[ni] = make_frag(ph[lh], ph[2 + lh], ph[4 + lh], ph[6 + lh]);
            bl[ni] = make_frag(pl[lh], pl[2 + lh], pl[4 + lh], pl[6 + lh]);
        }
#pragma unroll
        for (int mi = 0; mi < 2; ++mi)
#pragma unroll
            for (int ni = 0; ni < 2; ++ni) {
                acc_hh[mi][ni] = __builtin_amdgcn_wmma_i32_16x16x64_iu8(
                    true, ah[mi], true, bh[ni], acc_hh[mi][ni], false, false);
                acc_mm[mi][ni] = __builtin_amdgcn_wmma_i32_16x16x64_iu8(
                    true, ah[mi], false, bl[ni], acc_mm[mi][ni], false, false);
                acc_mm[mi][ni] = __builtin_amdgcn_wmma_i32_16x16x64_iu8(
                    false, al[mi], true, bh[ni], acc_mm[mi][ni], false, false);
                acc_ll[mi][ni] = __builtin_amdgcn_wmma_i32_16x16x64_iu8(
                    false, al[mi], false, bl[ni], acc_ll[mi][ni], false, false);
            }
    };

    const int NSTEP = K_DIM / 64;

#if HAVE_ASYNC
    async_copy(0, 0);
    async_copy(1, 64);
    wait_async_le6();
    __syncthreads();
    int cur = 0;
    for (int step = 0; step < NSTEP - 2; ++step) {
        int buf2 = cur + 2; if (buf2 >= 3) buf2 -= 3;
        async_copy(buf2, (step + 2) * 64);
        compute(cur);
        wait_async_le6();
        __syncthreads();
        ++cur; if (cur >= 3) cur -= 3;
    }
    compute(cur);
    wait_async_le0();
    __syncthreads();
    ++cur; if (cur >= 3) cur -= 3;
    compute(cur);
#else
    for (int step = 0; step < NSTEP; ++step) {
        const int kk = step * 64;
        u64 ra_h[4], ra_l[4], rb_h[2], rb_l[2];
#pragma unroll
        for (int i = 0; i < 4; ++i) {
            int lin = t + 256 * i;
            int row = lin >> 3, c8 = lin & 7;
            size_t off = (size_t)(m0 + row) * K_DIM + (size_t)kk + (size_t)c8 * 8;
            ra_h[i] = *reinterpret_cast<const u64*>(xh + off);
            ra_l[i] = *reinterpret_cast<const u64*>(xl + off);
        }
#pragma unroll
        for (int i = 0; i < 2; ++i) {
            int lin = t + 256 * i;
            int row = lin >> 3, c8 = lin & 7;
            size_t off = (size_t)(n0 + row) * K_DIM + (size_t)kk + (size_t)c8 * 8;
            rb_h[i] = *reinterpret_cast<const u64*>(wh + off);
            rb_l[i] = *reinterpret_cast<const u64*>(wl + off);
        }
        __syncthreads();
#pragma unroll
        for (int i = 0; i < 4; ++i) { int lin = t + 256 * i; sAh[0][lin] = ra_h[i]; sAl[0][lin] = ra_l[i]; }
#pragma unroll
        for (int i = 0; i < 2; ++i) { int lin = t + 256 * i; sBh[0][lin] = rb_h[i]; sBl[0][lin] = rb_l[i]; }
        __syncthreads();
        compute(0);
    }
#endif

#pragma unroll
    for (int ni = 0; ni < 2; ++ni) {
        const int ng = n0 + wn + 16 * ni + lr;
        const float qb = (float)quant_code(bias[ng]) * (1.0f / 256.0f);
#pragma unroll
        for (int mi = 0; mi < 2; ++mi) {
            const int mg0 = m0 + wm + 16 * mi + 8 * lh;
#pragma unroll
            for (int r = 0; r < 8; ++r) {
                long long s = ((long long)acc_hh[mi][ni][r] << 16)
                            + ((long long)acc_mm[mi][ni][r] << 8)
                            +  (long long)acc_ll[mi][ni][r];
                out[(size_t)(mg0 + r) * N_DIM + ng] = (float)s * (1.0f / 65536.0f) + qb;
            }
        }
    }
}

extern "C" void kernel_launch(void* const* d_in, const int* in_sizes, int n_in,
                              void* d_out, int out_size, void* d_ws, size_t ws_size,
                              hipStream_t stream) {
    const float* x = (const float*)d_in[0];
    const float* w = (const float*)d_in[1];
    const float* b = (const float*)d_in[2];
    float* out = (float*)d_out;

    unsigned char* ws = (unsigned char*)d_ws;
    const size_t xBytes = (size_t)M_DIM * K_DIM;
    const size_t wBytes = (size_t)N_DIM * K_DIM;
    unsigned char* xh = ws;
    unsigned char* xl = ws + xBytes;
    unsigned char* wh = ws + 2 * xBytes;
    unsigned char* wl = ws + 2 * xBytes + wBytes;

    {
        long long n4 = (long long)M_DIM * K_DIM / 4;
        quant_split_kernel<<<(unsigned)((n4 + 255) / 256), 256, 0, stream>>>(x, xh, xl, n4);
    }
    {
        long long n4 = (long long)N_DIM * K_DIM / 4;
        quant_split_kernel<<<(unsigned)((n4 + 255) / 256), 256, 0, stream>>>(w, wh, wl, n4);
    }
    dim3 grid(M_DIM / 128, N_DIM / 64);
    fpq_gemm_kernel<<<grid, 256, 0, stream>>>(xh, xl, wh, wl, b, out);
}
